// Head_2731599200830
// MI455X (gfx1250) — matrix operand order repaired and checked by reading, not compiled, not run on hardware
//
#include <hip/hip_runtime.h>

#define B_ 4
#define T_ 4096
#define C_ 512
#define H_ 64

typedef __attribute__((ext_vector_type(16))) __bf16 v16bf;
typedef __attribute__((ext_vector_type(8)))  __bf16 v8bf;
typedef __attribute__((ext_vector_type(8)))  float  v8f;

// ws layout (bf16 elems): [ Q : B*T*H ][ K : B*T*H ][ Vt : B*H*T ][ WB : 3*16*4*32*16 ]
// K-tile overreads land in Vt; Vt row overreads land in WB (finite, masked by p=0).
#define MAT_ELEMS ((size_t)B_ * T_ * H_)
#define WB_PER_MAT (16 * 4 * 32 * 16)   // kstep, ntile, lane, e

__device__ __forceinline__ v16bf concat8(v8bf lo, v8bf hi) {
  return __builtin_shufflevector(lo, hi, 0, 1, 2, 3, 4, 5, 6, 7,
                                 8, 9, 10, 11, 12, 13, 14, 15);
}

// ---------------------------------------------------------------------------
// Setup: W[C,H] f32 -> bf16 pre-permuted into WMMA B-fragment order.
// ---------------------------------------------------------------------------
__global__ __launch_bounds__(256) void pack_w_bf16(
    const float* __restrict__ Wq, const float* __restrict__ Wk,
    const float* __restrict__ Wv, __bf16* __restrict__ WB) {
  const int idx = blockIdx.x * 256 + threadIdx.x;      // 6144 jobs of 16 elems
  const int lane = idx & 31;
  int r = idx >> 5;
  const int nt = r & 3;  r >>= 2;
  const int ks = r & 15; r >>= 4;
  const int mat = r;                                   // 0..2
  const float* W = (mat == 0) ? Wq : (mat == 1) ? Wk : Wv;
  const int half = lane >> 4, lcol = lane & 15;
  __bf16* dst = WB + (size_t)idx * 16;
  const float* src = W + (size_t)(ks * 32 + half * 16) * H_ + nt * 16 + lcol;
#pragma unroll
  for (int e = 0; e < 16; ++e) dst[e] = (__bf16)src[(size_t)e * H_];
}

// ---------------------------------------------------------------------------
// Kernel 1: one wave per 16-row tile computes q, k AND v (x read once).
// All 12 B fragments of a K-step are loaded before the 12 WMMAs so the
// scheduler can clause the loads and stagger the loadcnt waits.
// ---------------------------------------------------------------------------
__global__ __launch_bounds__(128) void proj_bf16_wmma(
    const float* __restrict__ x, const __bf16* __restrict__ WB,
    __bf16* __restrict__ Qb, __bf16* __restrict__ Kb, __bf16* __restrict__ Vt) {
  const int wave = threadIdx.x >> 5;
  const int lane = threadIdx.x & 31;
  const int half = lane >> 4;
  const int lcol = lane & 15;

  const int gw = blockIdx.x * 4 + wave;    // 0 .. B*T/16-1
  const int b  = gw >> 8;                  // / (T/16)
  const int m0 = (gw & 255) << 4;

  const float* xrow = x + ((size_t)b * T_ + (m0 + lcol)) * C_;

  v8f acc[3][4] = {};
  for (int ks = 0; ks < 16; ++ks) {
    const float* xp = xrow + ks * 32 + half * 16;
    const v8f f0 = *(const v8f*)(xp);
    const v8f f1 = *(const v8f*)(xp + 8);
    v16bf a;
#pragma unroll
    for (int e = 0; e < 8; ++e) { a[e] = (__bf16)f0[e]; a[e + 8] = (__bf16)f1[e]; }

    v16bf wb[12];
#pragma unroll
    for (int mat = 0; mat < 3; ++mat)
#pragma unroll
      for (int nt = 0; nt < 4; ++nt)
        wb[mat * 4 + nt] = *(const v16bf*)(
            WB + (size_t)mat * WB_PER_MAT +
            ((size_t)((ks * 4 + nt) * 32) + lane) * 16);
#pragma unroll
    for (int mat = 0; mat < 3; ++mat)
#pragma unroll
      for (int nt = 0; nt < 4; ++nt)
        acc[mat][nt] = __builtin_amdgcn_wmma_f32_16x16x32_bf16(
            false, a, false, wb[mat * 4 + nt], (short)0, acc[mat][nt],
            false, false);
  }

  __bf16* qdst = Qb + ((size_t)b * T_) * H_;
  __bf16* kdst = Kb + ((size_t)b * T_) * H_;
#pragma unroll
  for (int nt = 0; nt < 4; ++nt)
#pragma unroll
    for (int j = 0; j < 8; ++j) {
      const int row = m0 + j + half * 8;     // C/D layout
      qdst[(size_t)row * H_ + nt * 16 + lcol] = (__bf16)acc[0][nt][j];
      kdst[(size_t)row * H_ + nt * 16 + lcol] = (__bf16)acc[1][nt][j];
    }
  // v transposed: lane holds 8 consecutive rows of channel nt*16+lcol
#pragma unroll
  for (int nt = 0; nt < 4; ++nt) {
    v8bf vb;
#pragma unroll
    for (int j = 0; j < 8; ++j) vb[j] = (__bf16)acc[2][nt][j];
    *(v8bf*)(Vt + ((size_t)b * H_ + nt * 16 + lcol) * T_ + m0 + half * 8) = vb;
  }
}

// ---------------------------------------------------------------------------
// Kernel 2: fused causal flash attention, transposed pipeline + 4-way split-K.
// One 16-row tile per 4-wave block; wave w handles key blocks 32w + 128k.
// Each wave keeps private (m, l, O^T); LDS merge at the end (wave 0 stores).
// ---------------------------------------------------------------------------
__global__ __launch_bounds__(128) void flash_attn_wmma(
    const __bf16* __restrict__ Qb, const __bf16* __restrict__ Kb,
    const __bf16* __restrict__ Vt, float* __restrict__ out) {
  __shared__ float ldsM[3][32];
  __shared__ float ldsL[3][32];
  __shared__ float ldsO[3][32][32];

  const int wave = threadIdx.x >> 5;
  const int lane = threadIdx.x & 31;
  const int half = lane >> 4;
  const int lcol = lane & 15;

  const int tile = blockIdx.x;             // 0 .. B*T/16-1
  const int b  = tile >> 8;
  const int m0 = (tile & 255) << 4;

  const __bf16* Qp = Qb + ((size_t)b * T_) * H_;
  const __bf16* Kp = Kb + ((size_t)b * T_) * H_;
  const __bf16* Vp = Vt + ((size_t)b * H_) * T_;

  // Q^T B-fragments (hoisted): lane = query row, contiguous channel chunk
  v16bf qb[2];
  {
    const __bf16* qrow = Qp + (size_t)(m0 + lcol) * H_;
#pragma unroll
    for (int kb = 0; kb < 2; ++kb)
      qb[kb] = *(const v16bf*)(qrow + kb * 32 + half * 16);
  }

  float mrun = -3.0e38f, lrun = 0.0f;
  v8f oT[4] = {};

  const float scale = 0.044194173824159216f;   // 512 ** -0.5 (per reference)
  const int row  = m0 + lcol;                  // this lane's query row
  const int sEnd = m0 + 16;

  for (int s0 = wave * 32; s0 < sEnd; s0 += 128) {
    // ---- S^T = K Q^T : two 16-key tiles, 2 K-steps each ----
    v8f sT[2] = {};
#pragma unroll
    for (int t = 0; t < 2; ++t) {
      const __bf16* krow = Kp + (size_t)(s0 + 16 * t + lcol) * H_;
#pragma unroll
      for (int kb = 0; kb < 2; ++kb) {
        const v8bf lo = *(const v8bf*)(krow + kb * 32 + half * 16);
        const v8bf hi = *(const v8bf*)(krow + kb * 32 + half * 16 + 8);
        sT[t] = __builtin_amdgcn_wmma_f32_16x16x32_bf16(
            false, concat8(lo, hi), false, qb[kb], (short)0, sT[t], false, false);
      }
    }

    // ---- issue V fragment loads early; softmax VALU hides their latency ----
    v8bf vlo[4], vhi[4];
#pragma unroll
    for (int nt = 0; nt < 4; ++nt) {
      const __bf16* vrow = Vp + (size_t)(nt * 16 + lcol) * T_ + s0;
      vlo[nt] = *(const v8bf*)(vrow + half * 16);
      vhi[nt] = *(const v8bf*)(vrow + half * 16 + 8);
    }

    // ---- scale + causal mask + online softmax (in-lane + 1 half-swap) ----
    float p[2][8];
    float mx = -3.0e38f;
#pragma unroll
    for (int t = 0; t < 2; ++t)
#pragma unroll
      for (int j = 0; j < 8; ++j) {
        const int key = s0 + 16 * t + 8 * half + j;   // D row = key-in-tile
        float sv = sT[t][j] * scale;
        if (key > row) sv = -3.0e38f;
        p[t][j] = sv;
        mx = fmaxf(mx, sv);
      }
    mx = fmaxf(mx, __shfl_xor(mx, 16, 32));           // merge lane halves
    const float mnew  = fmaxf(mrun, mx);
    const float alpha = __expf(mrun - mnew);
    mrun = mnew;

    float ls = 0.0f;
#pragma unroll
    for (int t = 0; t < 2; ++t)
#pragma unroll
      for (int j = 0; j < 8; ++j) {
        p[t][j] = __expf(p[t][j] - mnew);
        ls += p[t][j];
      }
    ls += __shfl_xor(ls, 16, 32);
    lrun = lrun * alpha + ls;

    // ---- build P^T B-fragment: one cross-half exchange, no LDS ----
    float recv[8];
#pragma unroll
    for (int j = 0; j < 8; ++j) {
      const float send = half ? p[0][j] : p[1][j];    // p[1-half][j]
      recv[j] = __shfl_xor(send, 16, 32);             // partner's p[half][j]
    }
    v16bf pb;
#pragma unroll
    for (int j = 0; j < 8; ++j) {
      pb[j]     = (__bf16)(half ? recv[j] : p[0][j]);
      pb[j + 8] = (__bf16)(half ? p[1][j] : recv[j]);
    }

    // ---- rescale O^T, then O^T += V^T P^T ----
#pragma unroll
    for (int nt = 0; nt < 4; ++nt)
#pragma unroll
      for (int j = 0; j < 8; ++j) oT[nt][j] *= alpha;

#pragma unroll
    for (int nt = 0; nt < 4; ++nt)
      oT[nt] = __builtin_amdgcn_wmma_f32_16x16x32_bf16(
          false, concat8(vlo[nt], vhi[nt]), false, pb, (short)0, oT[nt],
          false, false);
  }

  // ---- split-K merge: waves 1..3 publish, wave 0 combines and stores ----
  if (wave != 0) {
    ldsM[wave - 1][lane] = mrun;
    ldsL[wave - 1][lane] = lrun;
#pragma unroll
    for (int nt = 0; nt < 4; ++nt)
#pragma unroll
      for (int j = 0; j < 8; ++j) ldsO[wave - 1][lane][nt * 8 + j] = oT[nt][j];
  }
  __syncthreads();

  if (wave == 0) {
    float mstar = mrun;
#pragma unroll
    for (int w = 0; w < 3; ++w) mstar = fmaxf(mstar, ldsM[w][lane]);
    const float a0 = __expf(mrun - mstar);
    float lstar = lrun * a0;
#pragma unroll
    for (int nt = 0; nt < 4; ++nt) oT[nt] = oT[nt] * a0;
#pragma unroll
    for (int w = 0; w < 3; ++w) {
      const float aw = __expf(ldsM[w][lane] - mstar);
      lstar += ldsL[w][lane] * aw;
#pragma unroll
      for (int nt = 0; nt < 4; ++nt)
#pragma unroll
        for (int j = 0; j < 8; ++j)
          oT[nt][j] += ldsO[w][lane][nt * 8 + j] * aw;
    }
    const float inv_l = 1.0f / lstar;
    float* obase = out + ((size_t)b * T_ + row) * H_;
#pragma unroll
    for (int nt = 0; nt < 4; ++nt) {
      const v8f r = oT[nt] * inv_l;
      *(v8f*)(obase + nt * 16 + half * 8) = r;
    }
  }
}

// ---------------------------------------------------------------------------
extern "C" void kernel_launch(void* const* d_in, const int* in_sizes, int n_in,
                              void* d_out, int out_size, void* d_ws, size_t ws_size,
                              hipStream_t stream) {
  const float* x  = (const float*)d_in[0];   // [B,T,C]
  const float* Wk = (const float*)d_in[1];   // [C,H]
  const float* Wq = (const float*)d_in[2];   // [C,H]
  const float* Wv = (const float*)d_in[3];   // [C,H]

  __bf16* Qb = (__bf16*)d_ws;                // 2 MB
  __bf16* Kb = Qb + MAT_ELEMS;               // 2 MB
  __bf16* Vt = Kb + MAT_ELEMS;               // 2 MB, [B,H,T]
  __bf16* WB = Vt + MAT_ELEMS;               // 192 KB (absorbs Vt overreads)

  pack_w_bf16<<<24, 256, 0, stream>>>(Wq, Wk, Wv, WB);
  // B*T/16 = 1024 waves, 4 per block -> 256 blocks; each wave does q+k+v
  proj_bf16_wmma<<<256, 128, 0, stream>>>(x, WB, Qb, Kb, Vt);
  // one 16-row tile per 4-wave block, 4-way split over keys -> 1024 blocks
  flash_attn_wmma<<<1024, 128, 0, stream>>>(Qb, Kb, Vt, (float*)d_out);
}
